// GATLayer_88072599371930
// MI455X (gfx1250) — compile-verified, matrix operand order checked by reading, not run on hardware
//
#include <hip/hip_runtime.h>
#include <hip/hip_bf16.h>

// ---------------------------------------------------------------------------
// GAT layer for MI455X (gfx1250, wave32, WMMA + async-to-LDS).
// H=8 heads, B=2, N=2048 nodes, F=256 in-feat, O=64 out-feat/head.
// Kernel 1: per-head projection h = x @ W (bf16 WMMA, f32 accum) + attention
//           scalars s_src/s_dst. Stores h transposed (bf16 [H][B][O][N]).
// Kernel 2: fused masked-softmax + aggregation out = softmax(score) @ h via
//           bf16 WMMA; h^T chunks staged in LDS with double-buffered
//           GLOBAL_LOAD_ASYNC_TO_LDS_B128 (ASYNCcnt), f32 softmax math.
// ---------------------------------------------------------------------------

constexpr int kH = 8;
constexpr int kB = 2;
constexpr int kN = 2048;
constexpr int kF = 256;
constexpr int kO = 64;

// Padded LDS row for staged h^T chunks: 32 data bf16 + pad -> 56 elems (112 B).
// 112 B keeps 16-B transfer alignment and 28-dword row stride => lanes 0..15
// hit banks (28*lane mod 64), all distinct: conflict-free ds_load_b128.
constexpr int kPadRow = 56;

typedef __attribute__((ext_vector_type(16))) __bf16 v16bf;
typedef __attribute__((ext_vector_type(8)))  __bf16 v8bf;
typedef __attribute__((ext_vector_type(8)))  float  v8f;

union AFrag { v16bf v; __bf16 e[16]; };
union BVec8 { v8bf  v; __bf16 e[8];  };
union CFrag { v8f   v; float  e[8];  };
union F8    { float4 q[2]; float f[8]; };

// Per-lane 16-byte async copy global -> LDS (ASYNCcnt-tracked).
// LDS operand = low 32 bits of the flat pointer (flat->LDS truncates to 32b).
__device__ __forceinline__ void async_copy16(const void* gptr, void* lds_ptr) {
    const unsigned lds = (unsigned)(size_t)lds_ptr;
    asm volatile("global_load_async_to_lds_b128 %0, %1, off"
                 :: "v"(lds), "v"(gptr) : "memory");
}

// ---------------------------------------------------------------------------
// Kernel 1: projection + attention scalars.
// grid = (N/64, H*B), block = 128 (4 waves). Wave w computes rows
// [row0+16w, row0+16w+16) x all 64 output cols via 4 WMMA accumulators.
// ---------------------------------------------------------------------------
__global__ __launch_bounds__(128)
void gat_project(const float* __restrict__ x, const float* __restrict__ W,
                 const float* __restrict__ a_src, const float* __restrict__ a_dst,
                 unsigned short* __restrict__ hT_u16,
                 float* __restrict__ s_src, float* __restrict__ s_dst)
{
    __shared__ __bf16 sWt[kO][kF + 8];   // W^T per head, padded vs bank conflicts
    __shared__ float  sH[64][kO + 1];    // f32 h tile for the scalar reduction
    __shared__ float  sA[kO], sD[kO];

    const int head = blockIdx.y / kB;
    const int b    = blockIdx.y % kB;
    const int row0 = blockIdx.x * 64;
    const int tid  = threadIdx.x;
    const int lane = tid & 31;
    const int wave = tid >> 5;

    // Stage W^T (bf16) into LDS: sWt[o][f] = W[head][f][o]
    const float* Wh = W + (size_t)head * kF * kO;
    for (int i = tid; i < kF * kO; i += 128) {
        const int o = i >> 8;          // i / kF
        const int f = i & (kF - 1);    // i % kF
        sWt[o][f] = (__bf16)Wh[f * kO + o];
    }
    if (tid < kO) {
        sA[tid] = a_src[head * kO + tid];
        sD[tid] = a_dst[head * kO + tid];
    }
    __syncthreads();

    const int mrow = lane & 15;        // A-matrix row owned by this lane
    const int koff = (lane >> 4) * 8;  // K offset per half-wave (ISA A layout)
    const int row  = row0 + wave * 16 + mrow;
    const float* xrow = x + ((size_t)b * kN + row) * kF;

    CFrag c[4];
    #pragma unroll
    for (int g = 0; g < 4; ++g)
        #pragma unroll
        for (int i = 0; i < 8; ++i) c[g].e[i] = 0.0f;

    for (int kb = 0; kb < kF; kb += 32) {
        // A fragment: x row, f32 -> bf16. Lane covers K = kb+koff+{0..7,16..23}.
        F8 xlo, xhi;
        xlo.q[0] = *(const float4*)(xrow + kb + koff);
        xlo.q[1] = *(const float4*)(xrow + kb + koff + 4);
        xhi.q[0] = *(const float4*)(xrow + kb + koff + 16);
        xhi.q[1] = *(const float4*)(xrow + kb + koff + 20);
        AFrag a;
        #pragma unroll
        for (int i = 0; i < 8; ++i) {
            a.e[i]     = (__bf16)xlo.f[i];
            a.e[i + 8] = (__bf16)xhi.f[i];
        }
        // 4 B fragments (O tiles) from LDS W^T; contiguous 16B reads.
        #pragma unroll
        for (int g = 0; g < 4; ++g) {
            const int ocol = g * 16 + mrow;
            BVec8 blo, bhi;
            blo.v = *(const v8bf*)&sWt[ocol][kb + koff];
            bhi.v = *(const v8bf*)&sWt[ocol][kb + koff + 16];
            const v16bf bv = __builtin_shufflevector(
                blo.v, bhi.v, 0, 1, 2, 3, 4, 5, 6, 7, 8, 9, 10, 11, 12, 13, 14, 15);
            c[g].v = __builtin_amdgcn_wmma_f32_16x16x32_bf16(
                false, a.v, false, bv, (short)0, c[g].v, false, false);
        }
    }

    // Epilogue: write h^T (bf16, contiguous rows) + stage f32 tile in LDS.
    const size_t hTbase = ((size_t)(head * kB + b)) * kO * kN;
    const int mbase = wave * 16 + (lane >> 4) * 8;   // C rows: mbase + r
    #pragma unroll
    for (int g = 0; g < 4; ++g) {
        const int ocol = g * 16 + mrow;
        BVec8 st;
        #pragma unroll
        for (int r = 0; r < 8; ++r) {
            const float v = c[g].e[r];
            sH[mbase + r][ocol] = v;
            st.e[r] = (__bf16)v;
        }
        *(v8bf*)((__bf16*)hT_u16 + hTbase + (size_t)ocol * kN + row0 + mbase) = st.v;
    }
    __syncthreads();

    // Attention scalars: s = h . a  (one row per thread, 64 rows).
    if (tid < 64) {
        float accS = 0.0f, accD = 0.0f;
        #pragma unroll 8
        for (int o = 0; o < kO; ++o) {
            const float v = sH[tid][o];
            accS += v * sA[o];
            accD += v * sD[o];
        }
        const size_t idx = ((size_t)(head * kB + b)) * kN + row0 + tid;
        s_src[idx] = accS;
        s_dst[idx] = accD;
    }
}

// ---------------------------------------------------------------------------
// Kernel 2: fused masked softmax + WMMA aggregation + ReLU + head concat.
// grid = (N/64, H*B), block = 128. Two-pass softmax (adj is L2-resident after
// the first head: 33.5 MB << 192 MB L2). h^T B-operand chunks are staged once
// per block via double-buffered async-to-LDS (4x traffic reduction vs
// per-wave global loads) and overlapped with the exp/WMMA work.
// ---------------------------------------------------------------------------
__global__ __launch_bounds__(128)
void gat_attend(const float* __restrict__ adj,
                const unsigned short* __restrict__ hT_u16,
                const float* __restrict__ s_src_g, const float* __restrict__ s_dst_g,
                float* __restrict__ out)
{
    __shared__ float  sSrc[kN];                 // 8 KB: s_src for all columns
    __shared__ float  sDst[64];
    __shared__ float  sMax[64];
    __shared__ float  sDen[64];
    __shared__ __bf16 sHT[2][kO][kPadRow];      // 2 x 7 KB staged h^T chunks

    const int head = blockIdx.y / kB;
    const int b    = blockIdx.y % kB;
    const int row0 = blockIdx.x * 64;
    const int tid  = threadIdx.x;
    const int lane = tid & 31;
    const int wave = tid >> 5;

    const size_t sbase = ((size_t)(head * kB + b)) * kN;
    for (int j = tid; j < kN; j += 128) sSrc[j] = s_src_g[sbase + j];
    if (tid < 64) sDst[tid] = s_dst_g[sbase + row0 + tid];
    __syncthreads();

    const float* adjb = adj + (size_t)b * kN * kN;

    // ---- pass 1: per-row max of masked leaky-relu scores ----
    for (int rr = 0; rr < 16; ++rr) {
        const int   ml  = wave * 16 + rr;
        const float sd  = sDst[ml];
        const float* ar = adjb + (size_t)(row0 + ml) * kN;
        float mx = -3.0e38f;
        for (int j = lane * 4; j < kN; j += 128) {
            const float4 av = *(const float4*)(ar + j);
            const float4 sv = *(const float4*)(sSrc + j);
            float sc;
            if (av.x > 0.f) { sc = sd + sv.x; sc = sc >= 0.f ? sc : 0.2f * sc; mx = fmaxf(mx, sc); }
            if (av.y > 0.f) { sc = sd + sv.y; sc = sc >= 0.f ? sc : 0.2f * sc; mx = fmaxf(mx, sc); }
            if (av.z > 0.f) { sc = sd + sv.z; sc = sc >= 0.f ? sc : 0.2f * sc; mx = fmaxf(mx, sc); }
            if (av.w > 0.f) { sc = sd + sv.w; sc = sc >= 0.f ? sc : 0.2f * sc; mx = fmaxf(mx, sc); }
        }
        #pragma unroll
        for (int off = 16; off > 0; off >>= 1)
            mx = fmaxf(mx, __shfl_xor(mx, off, 32));
        if (lane == 0) sMax[ml] = mx;
    }
    __syncthreads();

    // ---- pass 2: softmax numerators as WMMA A-fragments, aggregate vs h^T ----
    const int mrow = lane & 15;
    const int koff = (lane >> 4) * 8;
    const int am   = wave * 16 + mrow;             // local A-row of this lane
    const float rmax = sMax[am];
    const float sd   = sDst[am];
    const float* arow = adjb + (size_t)(row0 + am) * kN;
    const __bf16* hT  = (const __bf16*)hT_u16 + ((size_t)(head * kB + b)) * kO * kN;

    CFrag c[4];
    #pragma unroll
    for (int g = 0; g < 4; ++g)
        #pragma unroll
        for (int i = 0; i < 8; ++i) c[g].e[i] = 0.0f;
    float den = 0.0f;

    // Cooperative async stage of one 32(j) x 64(o) bf16 chunk of h^T:
    // 256 x 16-byte transfers, 2 per thread (= 2 async instructions per wave).
    auto stage_chunk = [&](int jc, int buf) {
        #pragma unroll
        for (int k = 0; k < 2; ++k) {
            const int i = tid + k * 128;           // 0..255
            const int o = i >> 2;
            const int q = (i & 3) * 8;             // bf16 offset inside chunk row
            async_copy16(hT + (size_t)o * kN + jc + q, &sHT[buf][o][q]);
        }
    };

    stage_chunk(0, 0);                             // prologue fill of buffer 0

    for (int jc = 0; jc < kN; jc += 32) {
        const int cur = (jc >> 5) & 1;
        const bool more = (jc + 32) < kN;
        if (more) stage_chunk(jc + 32, cur ^ 1);   // prefetch next chunk
        // Wait for current chunk (leave the 2 just-issued ops in flight).
        if (more) asm volatile("s_wait_asynccnt 2" ::: "memory");
        else      asm volatile("s_wait_asynccnt 0" ::: "memory");
        __syncthreads();
        __builtin_prefetch(arow + jc + 128, 0, 0); // global_prefetch_b8 (adj)

        // A fragment: softmax numerators p = exp(score - rowmax), masked.
        AFrag a;
        #pragma unroll
        for (int half = 0; half < 2; ++half) {
            const int jb = jc + koff + half * 16;
            F8 av, sv;
            av.q[0] = *(const float4*)(arow + jb);
            av.q[1] = *(const float4*)(arow + jb + 4);
            sv.q[0] = *(const float4*)(sSrc + jb);
            sv.q[1] = *(const float4*)(sSrc + jb + 4);
            #pragma unroll
            for (int i = 0; i < 8; ++i) {
                float p = 0.0f;
                if (av.f[i] > 0.0f) {
                    float sc = sd + sv.f[i];
                    sc = sc >= 0.0f ? sc : 0.2f * sc;        // leaky_relu(0.2)
                    p = __expf(sc - rmax);
                }
                den += p;
                a.e[half * 8 + i] = (__bf16)p;
            }
        }
        // B fragments from the staged LDS chunk (conflict-free ds_load_b128).
        #pragma unroll
        for (int g = 0; g < 4; ++g) {
            const int ocol = g * 16 + mrow;
            BVec8 blo, bhi;
            blo.v = *(const v8bf*)&sHT[cur][ocol][koff];
            bhi.v = *(const v8bf*)&sHT[cur][ocol][koff + 16];
            const v16bf bv = __builtin_shufflevector(
                blo.v, bhi.v, 0, 1, 2, 3, 4, 5, 6, 7, 8, 9, 10, 11, 12, 13, 14, 15);
            c[g].v = __builtin_amdgcn_wmma_f32_16x16x32_bf16(
                false, a.v, false, bv, (short)0, c[g].v, false, false);
        }
        __syncthreads();   // all waves done with buf[cur] before it is refilled
    }

    // Row denominators: lanes L and L^16 cover complementary K halves of row.
    den += __shfl_xor(den, 16, 32);
    if (lane < 16) sDen[wave * 16 + lane] = den;
    __syncthreads();

    // Epilogue: divide, ReLU, scatter into head-concat layout [B][N][H*O].
    const int mbase = wave * 16 + (lane >> 4) * 8;
    float* outb = out + ((size_t)b * kN + row0) * (kH * kO) + head * kO;
    #pragma unroll
    for (int g = 0; g < 4; ++g) {
        const int ocol = g * 16 + mrow;
        #pragma unroll
        for (int r = 0; r < 8; ++r) {
            const int mloc = mbase + r;
            float v = c[g].e[r] / sDen[mloc];
            v = v > 0.0f ? v : 0.0f;
            outb[(size_t)mloc * (kH * kO) + ocol] = v;
        }
    }
}

// ---------------------------------------------------------------------------
extern "C" void kernel_launch(void* const* d_in, const int* in_sizes, int n_in,
                              void* d_out, int out_size, void* d_ws, size_t ws_size,
                              hipStream_t stream) {
    const float* x     = (const float*)d_in[0];
    const float* adj   = (const float*)d_in[1];
    const float* W     = (const float*)d_in[2];
    const float* a_src = (const float*)d_in[3];
    const float* a_dst = (const float*)d_in[4];
    float* out = (float*)d_out;

    // Workspace carve: h^T bf16 [H][B][O][N] (4 MB), s_src/s_dst f32 [H][B][N].
    unsigned short* hT = (unsigned short*)d_ws;
    const size_t hT_elems = (size_t)kH * kB * kO * kN;
    float* s_src = (float*)((char*)d_ws + hT_elems * sizeof(unsigned short));
    float* s_dst = s_src + (size_t)kH * kB * kN;

    dim3 grid(kN / 64, kH * kB);
    dim3 block(128);
    gat_project<<<grid, block, 0, stream>>>(x, W, a_src, a_dst, hT, s_src, s_dst);
    gat_attend<<<grid, block, 0, stream>>>(adj, hT, s_src, s_dst, out);
}
